// QuantizedLinear_43894565765814
// MI455X (gfx1250) — matrix operand order repaired and checked by reading, not compiled, not run on hardware
//
#include <hip/hip_runtime.h>

#define IN_F  8192
#define OUT_F 8192
#define BATCH 64

#define KCHUNK  256                 // K-halfs staged in LDS per buffer per chunk
#define ASTRIDE (KCHUNK + 8)        // padded row stride (halfs): conflict-free b128
#define SLAB    (64 * ASTRIDE)      // one A buffer (halfs)

typedef __attribute__((ext_vector_type(16))) _Float16 v16h;
typedef __attribute__((ext_vector_type(8)))  _Float16 v8h;
typedef __attribute__((ext_vector_type(8)))  float    v8f;
typedef __attribute__((ext_vector_type(4)))  int      v4i;

#if defined(__gfx1250__) && \
    __has_builtin(__builtin_amdgcn_global_load_async_to_lds_b128) && \
    __has_builtin(__builtin_amdgcn_s_wait_asynccnt)
#define HAVE_ASYNC_LDS 1
typedef __attribute__((address_space(1))) v4i as1_v4i;   // global v4i
typedef __attribute__((address_space(3))) v4i as3_v4i;   // LDS v4i
#endif

// ---------------------------------------------------------------------------
// Kernel 1: convert x (fp32) -> fp16 staging buffer, and per-row sums
// ---------------------------------------------------------------------------
__global__ void __launch_bounds__(256)
xcvt_rowsum_kernel(const float* __restrict__ x,
                   _Float16* __restrict__ xh,
                   float* __restrict__ rowsum) {
  __shared__ float red[256];
  const int b   = blockIdx.x;
  const int tid = threadIdx.x;
  const float*  xr = x  + (size_t)b * IN_F;
  _Float16*     xo = xh + (size_t)b * IN_F;
  float s = 0.0f;
  for (int j = tid; j < IN_F; j += 256) {
    float v = xr[j];
    s += v;
    xo[j] = (_Float16)v;
  }
  red[tid] = s;
  __syncthreads();
  for (int off = 128; off > 0; off >>= 1) {
    if (tid < off) red[tid] += red[tid + off];
    __syncthreads();
  }
  if (tid == 0) rowsum[b] = red[0];
}

// int32 weights (int8-valued) -> exact fp16 B fragment
__device__ __forceinline__ v16h cvt16(v4i q0, v4i q1, v4i q2, v4i q3) {
  v16h b;
#pragma unroll
  for (int e = 0; e < 4; ++e) {
    b[e]      = (_Float16)(float)q0[e];
    b[e + 4]  = (_Float16)(float)q1[e];
    b[e + 8]  = (_Float16)(float)q2[e];
    b[e + 12] = (_Float16)(float)q3[e];
  }
  return b;
}

// ---- chunk staging helpers: 64 rows x KCHUNK halfs, 8 x b128 per thread ----
#ifdef HAVE_ASYNC_LDS
__device__ __forceinline__ void stage_chunk_async(_Float16* dst,
                                                  const _Float16* xh, int kc) {
  const int c8 = threadIdx.x & 31;   // 16B column unit
  const int r0 = threadIdx.x >> 5;   // starting row
#pragma unroll
  for (int u = 0; u < 8; ++u) {
    const int row = r0 + 8 * u;
    __builtin_amdgcn_global_load_async_to_lds_b128(
        (as1_v4i*)(v4i*)(xh + (size_t)row * IN_F + kc + c8 * 8),
        (as3_v4i*)(v4i*)(dst + row * ASTRIDE + c8 * 8),
        0, 0);
  }
}
#else
struct StageRegs { v8h r[8]; };
__device__ __forceinline__ void stage_load(StageRegs& s,
                                           const _Float16* xh, int kc) {
  const int c8 = threadIdx.x & 31;
  const int r0 = threadIdx.x >> 5;
#pragma unroll
  for (int u = 0; u < 8; ++u)
    s.r[u] = *(const v8h*)(xh + (size_t)(r0 + 8 * u) * IN_F + kc + c8 * 8);
}
__device__ __forceinline__ void stage_store(const StageRegs& s, _Float16* dst) {
  const int c8 = threadIdx.x & 31;
  const int r0 = threadIdx.x >> 5;
#pragma unroll
  for (int u = 0; u < 8; ++u)
    *(v8h*)(dst + (r0 + 8 * u) * ASTRIDE + c8 * 8) = s.r[u];
}
#endif

// ---------------------------------------------------------------------------
// Kernel 2: WMMA GEMM.
//  * wave  = one 16-wide output-column tile x one K-split slice, full batch
//            (M=64 -> 4 accumulators): each weight streamed from HBM once
//  * block = 8 waves sharing the same K-slice; A slab double-buffered in LDS
//            (2 x 33KB), staging overlapped with compute (async-to-LDS when
//            available, register-staged otherwise)
//  * B     = depth-2 register pipeline (qa/qb) on the int32 weight stream
// ---------------------------------------------------------------------------
__global__ void __launch_bounds__(256)
qlinear_wmma_kernel(const int* __restrict__ wq,
                    const _Float16* __restrict__ xh,
                    float* __restrict__ part,
                    int nsplit) {
  __shared__ _Float16 atile[2 * SLAB];       // 67,584 bytes

  const int lane   = threadIdx.x & 31;
  const int wid    = (int)((blockIdx.x * 256 + threadIdx.x) >> 5);
  const int otile  = wid & 511;              // 8192/16 column tiles
  const int ks     = wid >> 9;               // K-split index (same for block)
  const int kper   = IN_F / nsplit;
  const int kbeg   = ks * kper;
  const int kend   = kbeg + kper;
  const int klast  = kend - 32;

  const int l15    = lane & 15;
  const int lanehi = lane >> 4;
  const int obase  = otile << 4;

  const int* wrow = wq + (size_t)(obase + l15) * IN_F + lanehi * 8;

  // LDS read bases (halfs) for the 4 M-tiles of A
  const int abase0 = (0 * 16 + l15) * ASTRIDE + lanehi * 8;
  const int abase1 = (1 * 16 + l15) * ASTRIDE + lanehi * 8;
  const int abase2 = (2 * 16 + l15) * ASTRIDE + lanehi * 8;
  const int abase3 = (3 * 16 + l15) * ASTRIDE + lanehi * 8;

  v8f c0 = {}, c1 = {}, c2 = {}, c3 = {};

  // ---- depth-2 prefetch of the weight stream ----
  v4i qa0 = *(const v4i*)(wrow + kbeg);
  v4i qa1 = *(const v4i*)(wrow + kbeg + 4);
  v4i qa2 = *(const v4i*)(wrow + kbeg + 16);
  v4i qa3 = *(const v4i*)(wrow + kbeg + 20);
  const int kb1 = (kbeg + 32 <= klast) ? kbeg + 32 : klast;
  v4i qb0 = *(const v4i*)(wrow + kb1);
  v4i qb1 = *(const v4i*)(wrow + kb1 + 4);
  v4i qb2 = *(const v4i*)(wrow + kb1 + 16);
  v4i qb3 = *(const v4i*)(wrow + kb1 + 20);

  // ---- prologue: stage first A chunk into buffer 0 ----
#ifdef HAVE_ASYNC_LDS
  stage_chunk_async(atile, xh, kbeg);
  __builtin_amdgcn_s_wait_asynccnt(0);
  __syncthreads();
#else
  StageRegs sreg;
  stage_load(sreg, xh, kbeg);
  stage_store(sreg, atile);
  __syncthreads();
#endif

  int p = 0;
  for (int kc = kbeg; kc < kend; kc += KCHUNK, p ^= 1) {
    const int kn = kc + KCHUNK;
    // overlap: start staging next chunk into the other buffer
#ifdef HAVE_ASYNC_LDS
    if (kn < kend) stage_chunk_async(atile + (p ^ 1) * SLAB, xh, kn);
#else
    if (kn < kend) stage_load(sreg, xh, kn);
#endif
    const _Float16* ab = atile + p * SLAB;

#pragma unroll
    for (int kk = 0; kk < KCHUNK; kk += 64) {
      // ================= step A: k = kc + kk =================
      {
        v16h bmat = cvt16(qa0, qa1, qa2, qa3);
        int kp = kc + kk + 64; if (kp > klast) kp = klast;
        qa0 = *(const v4i*)(wrow + kp);
        qa1 = *(const v4i*)(wrow + kp + 4);
        qa2 = *(const v4i*)(wrow + kp + 16);
        qa3 = *(const v4i*)(wrow + kp + 20);

        const int co = kk;
        union { v16h v; v8h h[2]; } a0, a1, a2, a3;
        a0.h[0] = *(const v8h*)(ab + abase0 + co);
        a0.h[1] = *(const v8h*)(ab + abase0 + co + 16);
        a1.h[0] = *(const v8h*)(ab + abase1 + co);
        a1.h[1] = *(const v8h*)(ab + abase1 + co + 16);
        a2.h[0] = *(const v8h*)(ab + abase2 + co);
        a2.h[1] = *(const v8h*)(ab + abase2 + co + 16);
        a3.h[0] = *(const v8h*)(ab + abase3 + co);
        a3.h[1] = *(const v8h*)(ab + abase3 + co + 16);

        c0 = __builtin_amdgcn_wmma_f32_16x16x32_f16(false, a0.v, false, bmat,
                                                    (short)0, c0, false, false);
        c1 = __builtin_amdgcn_wmma_f32_16x16x32_f16(false, a1.v, false, bmat,
                                                    (short)0, c1, false, false);
        c2 = __builtin_amdgcn_wmma_f32_16x16x32_f16(false, a2.v, false, bmat,
                                                    (short)0, c2, false, false);
        c3 = __builtin_amdgcn_wmma_f32_16x16x32_f16(false, a3.v, false, bmat,
                                                    (short)0, c3, false, false);
      }
      // ================= step B: k = kc + kk + 32 =================
      {
        v16h bmat = cvt16(qb0, qb1, qb2, qb3);
        int kp = kc + kk + 96; if (kp > klast) kp = klast;
        qb0 = *(const v4i*)(wrow + kp);
        qb1 = *(const v4i*)(wrow + kp + 4);
        qb2 = *(const v4i*)(wrow + kp + 16);
        qb3 = *(const v4i*)(wrow + kp + 20);

        const int co = kk + 32;
        union { v16h v; v8h h[2]; } a0, a1, a2, a3;
        a0.h[0] = *(const v8h*)(ab + abase0 + co);
        a0.h[1] = *(const v8h*)(ab + abase0 + co + 16);
        a1.h[0] = *(const v8h*)(ab + abase1 + co);
        a1.h[1] = *(const v8h*)(ab + abase1 + co + 16);
        a2.h[0] = *(const v8h*)(ab + abase2 + co);
        a2.h[1] = *(const v8h*)(ab + abase2 + co + 16);
        a3.h[0] = *(const v8h*)(ab + abase3 + co);
        a3.h[1] = *(const v8h*)(ab + abase3 + co + 16);

        c0 = __builtin_amdgcn_wmma_f32_16x16x32_f16(false, a0.v, false, bmat,
                                                    (short)0, c0, false, false);
        c1 = __builtin_amdgcn_wmma_f32_16x16x32_f16(false, a1.v, false, bmat,
                                                    (short)0, c1, false, false);
        c2 = __builtin_amdgcn_wmma_f32_16x16x32_f16(false, a2.v, false, bmat,
                                                    (short)0, c2, false, false);
        c3 = __builtin_amdgcn_wmma_f32_16x16x32_f16(false, a3.v, false, bmat,
                                                    (short)0, c3, false, false);
      }
    }

    // ---- publish next chunk ----
#ifdef HAVE_ASYNC_LDS
    __builtin_amdgcn_s_wait_asynccnt(0);
    __syncthreads();
#else
    __syncthreads();                           // block done reading buffers
    if (kn < kend) stage_store(sreg, atile + (p ^ 1) * SLAB);
    __syncthreads();
#endif
  }

  // ---- scatter raw-dot partials (disjoint per-split slabs: deterministic) --
  float* pout = part + (size_t)ks * BATCH * OUT_F;
  const int mhi = lanehi * 8;
#pragma unroll
  for (int r = 0; r < 8; ++r) {
    pout[(size_t)( 0 + mhi + r) * OUT_F + obase + l15] = c0[r];
    pout[(size_t)(16 + mhi + r) * OUT_F + obase + l15] = c1[r];
    pout[(size_t)(32 + mhi + r) * OUT_F + obase + l15] = c2[r];
    pout[(size_t)(48 + mhi + r) * OUT_F + obase + l15] = c3[r];
  }
}

// ---------------------------------------------------------------------------
// Kernel 3: combine K-split partials and apply scale / zero-point / bias.
// ---------------------------------------------------------------------------
__global__ void __launch_bounds__(256)
finalize_kernel(const float* __restrict__ part,
                const float* __restrict__ rowsum,
                const float* __restrict__ wscale,
                const float* __restrict__ wzp,
                const int*   __restrict__ bias_q,
                const float* __restrict__ bscale,
                const float* __restrict__ bzp,
                float* __restrict__ out,
                int nsplit) {
  const int idx = (int)(blockIdx.x * blockDim.x + threadIdx.x);
  if (idx >= BATCH * OUT_F) return;
  const int o = idx & (OUT_F - 1);
  const int b = idx >> 13;               // OUT_F == 2^13
  float acc = 0.0f;
  for (int s = 0; s < nsplit; ++s)
    acc += part[(size_t)s * BATCH * OUT_F + idx];
  const float bias = (float)bias_q[o] * bscale[o] + bzp[o];
  out[idx] = wscale[o] * acc + wzp[o] * rowsum[b] + bias;
}

// ---------------------------------------------------------------------------
extern "C" void kernel_launch(void* const* d_in, const int* in_sizes, int n_in,
                              void* d_out, int out_size, void* d_ws, size_t ws_size,
                              hipStream_t stream) {
  const float* x      = (const float*)d_in[0];
  const int*   wq     = (const int*)  d_in[1];
  const float* wscale = (const float*)d_in[2];
  const float* wzp    = (const float*)d_in[3];
  const int*   bias_q = (const int*)  d_in[4];
  const float* bscale = (const float*)d_in[5];
  const float* bzp    = (const float*)d_in[6];
  float*       out    = (float*)d_out;

  // workspace layout: [ xh fp16 1MB | rowsum (4KB slot) | partials nsplit*2MB ]
  char*     ws     = (char*)d_ws;
  _Float16* xh     = (_Float16*)ws;
  float*    rowsum = (float*)(ws + (size_t)BATCH * IN_F * 2);
  float*    part   = (float*)(ws + (size_t)BATCH * IN_F * 2 + 4096);

  const size_t base = (size_t)BATCH * IN_F * 2 + 4096;
  const size_t slab = (size_t)BATCH * OUT_F * 4;   // 2 MB per split
  int nsplit = 1;
  if      (ws_size >= base + 8 * slab) nsplit = 8;
  else if (ws_size >= base + 4 * slab) nsplit = 4;
  else if (ws_size >= base + 2 * slab) nsplit = 2;

  xcvt_rowsum_kernel<<<BATCH, 256, 0, stream>>>(x, xh, rowsum);

  const int waves  = (OUT_F / 16) * nsplit;        // 512 * nsplit
  const int blocks = waves / 8;                    // 8 waves (256 thr) / block
  qlinear_wmma_kernel<<<blocks, 256, 0, stream>>>(wq, xh, part, nsplit);

  finalize_kernel<<<(BATCH * OUT_F) / 256, 256, 0, stream>>>(
      part, rowsum, wscale, wzp, bias_q, bscale, bzp, out, nsplit);
}
